// Dense_test_1layer_40510131536139
// MI455X (gfx1250) — matrix operand order repaired and checked by reading, not compiled, not run on hardware
//
#include <hip/hip_runtime.h>
#include <math.h>

// ---------------------------------------------------------------------------
// Recurrent branch-sparse SNN (SHD), persistent per-batch-tile WMMA kernel.
//   B=256, T=250, D=700, N=200, BR=4, OUT=35, P=900.
// Each workgroup owns 16 batch rows for all 250 timesteps. Main GEMM per step:
// 16x928(f16) @ 928x800 via v_wmma_f32_16x16x32_f16 with ping-pong register
// double-buffering; x_{t+1} prefetched via GLOBAL_LOAD_ASYNC_TO_LDS_B128.
// Neuron state (dinp, mem1) lives in LDS; K-loop unrolling is clamped so the
// prefetched fragments stay in registers (no scratch spills in the hot loop).
// ---------------------------------------------------------------------------

typedef _Float16 v16h __attribute__((ext_vector_type(16)));
typedef _Float16 v8h  __attribute__((ext_vector_type(8)));
typedef float    v8f  __attribute__((ext_vector_type(8)));
typedef int      v4i_g __attribute__((vector_size(16)));   // matches builtin param type

#define B_    256
#define T_    250
#define D_    700
#define N_    200
#define BR_   4
#define OUT_  35
#define P_    900
#define KP    928        // padded K: [0,700)=x, [700,704)=0, [704,904)=spk, [904,928)=0
#define SPK0  704        // 704 = 22*32 -> readout GEMM uses K-tiles 22..28
#define NKT   (KP / 32)  // 29 K-tiles
#define NBCOL 800        // N*BR hidden columns
#define KO    224        // padded readout K (200 -> 224 = 7*32)
#define OCOL  48         // padded readout cols (35 -> 48 = 3*16)
#define VTH_  1.0f

#if defined(__has_builtin)
#if __has_builtin(__builtin_amdgcn_global_load_async_to_lds_b128)
#define HAVE_ASYNC_LDS 1
#endif
#endif

__device__ __forceinline__ float sigmoidf_(float x) { return 1.0f / (1.0f + __expf(-x)); }

__device__ __forceinline__ void wait_asynccnt0() {
#if defined(HAVE_ASYNC_LDS)
#if __has_builtin(__builtin_amdgcn_s_wait_asynccnt)
  __builtin_amdgcn_s_wait_asynccnt(0);
#else
  asm volatile("s_wait_asynccnt 0x0" ::: "memory");
#endif
#endif
}

// ---- prep: Wt[col][kk] = f16((W1*mask)[col][k(kk)]), column-major, K padded ----
__global__ void prep_w1_kernel(const float* __restrict__ W1, const float* __restrict__ mask,
                               _Float16* __restrict__ Wt) {
  int idx = blockIdx.x * blockDim.x + threadIdx.x;
  if (idx >= NBCOL * KP) return;
  int j = idx / KP, kk = idx - j * KP;
  int ks = -1;
  if (kk < D_) ks = kk;                                          // feedforward part
  else if (kk >= SPK0 && kk < SPK0 + N_) ks = D_ + (kk - SPK0);  // recurrent part
  float v = 0.f;
  if (ks >= 0) v = W1[j * P_ + ks] * mask[j * P_ + ks];
  Wt[j * KP + kk] = (_Float16)v;
}

// ---- prep: W2t[col][kk] column-major, padded to 48 x 224 ----
__global__ void prep_w2_kernel(const float* __restrict__ W2, _Float16* __restrict__ W2t) {
  int idx = blockIdx.x * blockDim.x + threadIdx.x;
  if (idx >= OCOL * KO) return;
  int o = idx / KO, kk = idx - o * KO;
  float v = (o < OUT_ && kk < N_) ? W2[o * N_ + kk] : 0.f;
  W2t[o * KO + kk] = (_Float16)v;
}

// Load A fragment (16x32 f16) from LDS k-matrix, row-major stride KP.
// 16-bit A layout: lanes 0-15 M=row, elems 0..15 -> K=kt*32+0..15;
// lanes 16-31 get K offset +16 (contiguous, matches B fragments).
__device__ __forceinline__ v16h load_a_frag(const _Float16* kbase, int row, int kt, int half) {
  const _Float16* p = kbase + row * KP + kt * 32 + half * 16;
  v8h lo = *(const v8h*)p;
  v8h hi = *(const v8h*)(p + 8);
  v16h a;
#pragma unroll
  for (int i = 0; i < 8; ++i) { a[i] = lo[i]; a[i + 8] = hi[i]; }
  return a;
}

__launch_bounds__(320, 1)
__global__ void snn_persistent_kernel(const float* __restrict__ x,
                                      const float* __restrict__ b1,
                                      const float* __restrict__ tau_m1,
                                      const float* __restrict__ tau_n1,
                                      const float* __restrict__ b2,
                                      const float* __restrict__ tau_m2,
                                      const _Float16* __restrict__ Wt,
                                      const _Float16* __restrict__ W2t,
                                      float* __restrict__ out) {
  __shared__ __align__(16) _Float16 k_lds[16 * KP];       // [x_t | pad | spikes | pad]
  __shared__ __align__(16) float x_stage[16 * D_];        // raw fp32 x tile (async)
  __shared__ __align__(16) float dinp_lds[16 * NBCOL];    // dendritic state, per-lane chunks
  __shared__ __align__(16) float mem1_lds[N_ * 16];       // membrane state [neuron][row]
  __shared__ float out_lds[16 * OCOL];

  const int tid  = threadIdx.x;
  const int lane = tid & 31;
  const int w    = tid >> 5;     // wave 0..9; wave w owns hidden cols [80w, 80w+80)
  const int half = lane >> 4;    // K-half selector for WMMA fragments
  const int ln   = lane & 15;
  const int b0   = blockIdx.x * 16;

  // zero LDS once: pads stay zero forever, spike region = initial spk1 = 0
  for (int i = tid; i < 16 * KP; i += 320) k_lds[i] = (_Float16)0.f;
  for (int i = tid; i < 16 * NBCOL; i += 320) dinp_lds[i] = 0.f;
  for (int i = tid; i < N_ * 16; i += 320) mem1_lds[i] = 0.f;

  // per-lane tile constants (column j = 80w + 16c + ln; neuron = j>>2, branch = j&3)
  float betav[5], ombeta[5], b1v[5], a1v[5];
#pragma unroll
  for (int c = 0; c < 5; ++c) {
    int col = 80 * w + 16 * c + ln;
    betav[c]  = sigmoidf_(tau_n1[col]);   // tau_n1 is (N,BR) row-major == flat[col]
    ombeta[c] = 1.f - betav[c];
    b1v[c]    = b1[col];
    a1v[c]    = sigmoidf_(tau_m1[col >> 2]);
  }
  float a2v = 0.f, b2v = 0.f;
  if (w < 3) {
    int o = 16 * w + ln;
    if (o < OUT_) { a2v = sigmoidf_(tau_m2[o]); b2v = b2[o]; }
  }

  v8f mem2, msum;
#pragma unroll
  for (int i = 0; i < 8; ++i) { mem2[i] = 0.f; msum[i] = 0.f; }

  // issue async prefetch of x_t tile (fp32) into x_stage; 2800 x 16B chunks
  auto issue_x_async = [&](int t) {
#if defined(HAVE_ASYNC_LDS)
#pragma unroll 1
    for (int i = tid; i < 16 * (D_ / 4); i += 320) {   // D_/4 = 175 chunks per row
      int m = i / (D_ / 4);
      int d4 = (i - m * (D_ / 4)) * 4;
      const float* src = x + (size_t)(b0 + m) * T_ * D_ + (size_t)t * D_ + d4;
      __builtin_amdgcn_global_load_async_to_lds_b128(
          (__attribute__((address_space(1))) v4i_g*)src,
          (__attribute__((address_space(3))) v4i_g*)&x_stage[m * D_ + d4], 0, 0);
    }
#else
    (void)t;
#endif
  };

  const _Float16* wbase = Wt + (size_t)(80 * w + ln) * KP + 16 * half;  // tile stride 16*KP

#if defined(HAVE_ASYNC_LDS)
  issue_x_async(0);
#endif

#pragma unroll 1
  for (int t = 0; t < T_; ++t) {
    wait_asynccnt0();
    __syncthreads();  // B0: previous GEMM done before overwriting x region
#if defined(HAVE_ASYNC_LDS)
    // convert prefetched fp32 -> f16 into k_lds
#pragma unroll 1
    for (int i = tid; i < 16 * D_; i += 320) {
      int m = i / D_, d = i - m * D_;
      k_lds[m * KP + d] = (_Float16)x_stage[m * D_ + d];
    }
#else
#pragma unroll 1
    for (int i = tid; i < 16 * D_; i += 320) {
      int m = i / D_, d = i - m * D_;
      k_lds[m * KP + d] = (_Float16)x[(size_t)(b0 + m) * T_ * D_ + (size_t)t * D_ + d];
    }
#endif
    __syncthreads();  // B1: x staged; previous spikes already in place
#if defined(HAVE_ASYNC_LDS)
    if (t + 1 < T_) issue_x_async(t + 1);   // overlap next x fetch with this GEMM
#endif

    // ---- main GEMM: cur(16x800) = k(16x928) @ Wt, acc preloaded with b1 ----
    v8f acc[5];
#pragma unroll
    for (int c = 0; c < 5; ++c) {
#pragma unroll
      for (int i = 0; i < 8; ++i) acc[c][i] = b1v[c];
    }

    v16h a0, a1f, bb0[5], bb1[5];
    a0 = load_a_frag(k_lds, ln, 0, half);
#pragma unroll
    for (int c = 0; c < 5; ++c) bb0[c] = *(const v16h*)(wbase + (size_t)c * 16 * KP);

    auto gemm_step = [&](int kt, v16h& a_use, v16h (&b_use)[5],
                         int ktn, v16h& a_pre, v16h (&b_pre)[5]) {
      if (ktn < NKT) {   // prefetch next K-tile while this one computes
        a_pre = load_a_frag(k_lds, ln, ktn, half);
#pragma unroll
        for (int c = 0; c < 5; ++c)
          b_pre[c] = *(const v16h*)(wbase + (size_t)c * 16 * KP + ktn * 32);
      }
#pragma unroll
      for (int c = 0; c < 5; ++c)
        acc[c] = __builtin_amdgcn_wmma_f32_16x16x32_f16(false, a_use, false, b_use[c],
                                                        (short)0, acc[c], false, false);
    };

    // ping-pong double buffer; unroll clamped so fragments stay in registers
#pragma unroll 1
    for (int kt = 0; kt < NKT; kt += 2) {     // NKT = 29, odd -> last pair is lone
      gemm_step(kt,     a0,  bb0, kt + 1, a1f, bb1);
      if (kt + 1 < NKT) gemm_step(kt + 1, a1f, bb1, kt + 2, a0, bb0);
    }
    __syncthreads();  // B2: all waves finished reading old spikes

    // ---- dendritic filter, branch sum, LIF, spike writeback (state in LDS) ----
#pragma unroll
    for (int c = 0; c < 5; ++c) {
      float* dp = &dinp_lds[(size_t)(w * 160 + c * 32 + lane) * 8];   // 32B chunk per lane
      v8f d = *(v8f*)dp;
      d = betav[c] * d + ombeta[c] * acc[c];
      *(v8f*)dp = d;
      float ls[8];
#pragma unroll
      for (int i = 0; i < 8; ++i) {   // sum 4 branches = 4 adjacent lanes (butterfly)
        float s = d[i];
        s += __shfl_xor(s, 1, 32);
        s += __shfl_xor(s, 2, 32);
        ls[i] = s;
      }
      int neuron = (80 * w + 16 * c + ln) >> 2;   // shared by 4-lane group
      float m1[8], spknew[8];
#pragma unroll
      for (int i = 0; i < 8; ++i)      // broadcast read: same addr in 4-lane group
        m1[i] = mem1_lds[neuron * 16 + (i + 8 * half)];
#pragma unroll
      for (int i = 0; i < 8; ++i) {    // C layout: row m = i + 8*half
        float spk_old = (m1[i] - VTH_ > 0.f) ? 1.f : 0.f;  // recompute last spike
        float m = m1[i] * a1v[c] + (1.f - a1v[c]) * ls[i] - VTH_ * spk_old;
        m1[i]     = m;
        spknew[i] = (m - VTH_ > 0.f) ? 1.f : 0.f;
      }
      if ((lane & 3) == 0) {           // group leader writes state + spike
#pragma unroll
        for (int i = 0; i < 8; ++i) {
          mem1_lds[neuron * 16 + (i + 8 * half)] = m1[i];
          k_lds[(i + 8 * half) * KP + SPK0 + neuron] = (_Float16)spknew[i];
        }
      }
    }
    __syncthreads();  // B3: new spikes visible

    // ---- readout: mem2 = mem2*a2 + (1-a2)*(spk @ W2^T + b2), waves 0..2 ----
    if (w < 3) {
      v8f acc2;
#pragma unroll
      for (int i = 0; i < 8; ++i) acc2[i] = b2v;
#pragma unroll
      for (int kt = 22; kt < 29; ++kt) {   // spike region K-tiles (zero-padded)
        v16h a = load_a_frag(k_lds, ln, kt, half);
        const v16h bm = *(const v16h*)(W2t + (size_t)(16 * w + ln) * KO + (kt - 22) * 32 + 16 * half);
        acc2 = __builtin_amdgcn_wmma_f32_16x16x32_f16(false, a, false, bm,
                                                      (short)0, acc2, false, false);
      }
#pragma unroll
      for (int i = 0; i < 8; ++i) {
        mem2[i] = mem2[i] * a2v + (1.f - a2v) * acc2[i];
        if (t > 0) msum[i] += mem2[i];
      }
    }
  }

  // ---- epilogue: out = log_softmax(msum / T) ----
  if (w < 3) {
#pragma unroll
    for (int i = 0; i < 8; ++i)
      out_lds[(i + 8 * half) * OCOL + 16 * w + ln] = msum[i] * (1.f / (float)T_);
  }
  __syncthreads();
  if (tid < 16) {
    const float* r = &out_lds[tid * OCOL];
    float mx = -1e30f;
    for (int o = 0; o < OUT_; ++o) mx = fmaxf(mx, r[o]);
    float se = 0.f;
    for (int o = 0; o < OUT_; ++o) se += __expf(r[o] - mx);
    float lse = mx + __logf(se);
    for (int o = 0; o < OUT_; ++o) out[(size_t)(b0 + tid) * OUT_ + o] = r[o] - lse;
  }
}

extern "C" void kernel_launch(void* const* d_in, const int* in_sizes, int n_in,
                              void* d_out, int out_size, void* d_ws, size_t ws_size,
                              hipStream_t stream) {
  const float* x      = (const float*)d_in[0];
  const float* W1     = (const float*)d_in[1];
  const float* b1     = (const float*)d_in[2];
  const float* mask   = (const float*)d_in[3];
  const float* tau_m1 = (const float*)d_in[4];
  const float* tau_n1 = (const float*)d_in[5];
  const float* W2     = (const float*)d_in[6];
  const float* b2     = (const float*)d_in[7];
  const float* tau_m2 = (const float*)d_in[8];
  float* out = (float*)d_out;

  // workspace: Wt (800x928 f16 = 1.45MB) then W2t (48x224 f16 = 21.5KB)
  _Float16* Wt  = (_Float16*)d_ws;
  _Float16* W2t = (_Float16*)((char*)d_ws + (size_t)NBCOL * KP * sizeof(_Float16));

  {
    int total = NBCOL * KP;
    prep_w1_kernel<<<(total + 255) / 256, 256, 0, stream>>>(W1, mask, Wt);
  }
  {
    int total = OCOL * KO;
    prep_w2_kernel<<<(total + 255) / 256, 256, 0, stream>>>(W2, W2t);
  }
  // 16 workgroups (one per 16 batch rows) x 10 waves; persistent over all T
  snn_persistent_kernel<<<B_ / 16, 320, 0, stream>>>(x, b1, tau_m1, tau_n1, b2, tau_m2,
                                                     Wt, W2t, out);
}
